// EquivariantDGCNN_25993142075793
// MI455X (gfx1250) — matrix operand order repaired and checked by reading, not compiled, not run on hardware
//
#include <hip/hip_runtime.h>
#include <hip/hip_bf16.h>

typedef __attribute__((ext_vector_type(16))) _Float16 v16h;
typedef __attribute__((ext_vector_type(8)))  _Float16 v8h;
typedef __attribute__((ext_vector_type(8)))  float    v8f;
typedef __attribute__((ext_vector_type(2)))  float    v2f;

#define BQ   4
#define NP   4096
#define NPD  4100       // padded dist row stride (dwords): skews rows by 4 banks
#define KNNK 16

__device__ __forceinline__ float silu_f(float v) {
    return v / (1.0f + __expf(-v));
}

// Wave-private LDS ordering: DS ops from one wave are processed in order
// (ISA 7.3), so a compiler fence + scheduling barrier is sufficient for
// intra-wave LDS communication (no s_barrier needed).
__device__ __forceinline__ void wave_lds_fence() {
    __asm__ volatile("" ::: "memory");
    __builtin_amdgcn_wave_barrier();
    __asm__ volatile("" ::: "memory");
}

__device__ __forceinline__ v8f wmma_f16(v16h a, v16h b, v8f c) {
    // D = A(16x32 f16) * B(32x16 f16) + C(16x16 f32)
    return __builtin_amdgcn_wmma_f32_16x16x32_f16(false, a, false, b, (short)0, c, false, false);
}

__device__ __forceinline__ v8f wmma_f32x4(v2f a, v2f b, v8f c) {
    // D = A(16x4 f32) * B(4x16 f32) + C(16x16 f32)
    return __builtin_amdgcn_wmma_f32_16x16x4_f32(false, a, false, b, (short)0, c, false, false);
}

__device__ __forceinline__ v16h cat8(v8h lo, v8h hi) {
    return __builtin_shufflevector(lo, hi, 0, 1, 2, 3, 4, 5, 6, 7,
                                           8, 9, 10, 11, 12, 13, 14, 15);
}

// ---------------------------------------------------------------------------
// Kernel 0: pts [B,3,N] -> x0 [B,N,12] (zero padded), f0 [B,N,64] (=1 in col 0)
// ---------------------------------------------------------------------------
__global__ void egnn_init_kernel(const float* __restrict__ pts,
                                 float* __restrict__ x0,
                                 float* __restrict__ f0) {
    int t = blockIdx.x * blockDim.x + threadIdx.x;
    if (t >= BQ * NP) return;
    int b = t / NP, n = t % NP;
    float* xr = x0 + (size_t)(b * NP + n) * 12;
#pragma unroll
    for (int d = 0; d < 3; ++d) xr[d] = pts[(b * 3 + d) * NP + n];
#pragma unroll
    for (int d = 3; d < 12; ++d) xr[d] = 0.0f;
    float* fr = f0 + (size_t)(b * NP + n) * 64;
    fr[0] = 1.0f;
#pragma unroll
    for (int j = 1; j < 64; ++j) fr[j] = 0.0f;
}

// ---------------------------------------------------------------------------
// Kernel 1: KNN. Block = 256 threads (8 waves), 16 rows of one batch.
// Phase 1: f32 WMMA 16x16x4 gram -> padded LDS slab (bank-conflict-free).
// Phase 2: per-row top-(K+1) largest (f4 scans + shfl argmax), drop first.
// ---------------------------------------------------------------------------
template <int CC>
__global__ void egnn_knn_kernel(const float* __restrict__ x,
                                int* __restrict__ idx) {
    constexpr int NCH = (CC + 3) / 4;     // 1 (CC=3) or 3 (CC=12)
    extern __shared__ float lds[];
    float* dist = lds;                    // [16][NPD]
    float* sq   = dist + 16 * NPD;        // [NP]
    float* xr   = sq + NP;                // [16][12]
    float* sqr  = xr + 16 * 12;           // [16]

    int b   = blockIdx.x / (NP / 16);
    int r0  = (blockIdx.x % (NP / 16)) * 16;
    int tid = threadIdx.x;

    if (tid < 16 * 12) {
        int r = tid / 12, c = tid % 12;
        xr[tid] = x[(size_t)(b * NP + r0 + r) * 12 + c];   // zero-padded beyond CC
    }
    for (int m = tid; m < NP; m += 256) {
        const float* xm = x + (size_t)(b * NP + m) * 12;
        __builtin_prefetch(xm + 256 * 12, 0, 1);
        float s = 0.0f;
#pragma unroll
        for (int c = 0; c < CC; ++c) { float v = xm[c]; s += v * v; }
        sq[m] = s;
    }
    __syncthreads();
    if (tid < 16) {
        float s = 0.0f;
#pragma unroll
        for (int c = 0; c < CC; ++c) { float v = xr[tid * 12 + c]; s += v * v; }
        sqr[tid] = s;
    }
    __syncthreads();

    // ---- Phase 1: distance slab via f32 WMMA ----
    int lane = tid & 31, wv = tid >> 5;
    int Mr   = lane & 15;
    int kadd = (lane < 16) ? 0 : 2;       // f32 A/B: half-waves hold K {0,1} / {2,3}
    v2f a2[NCH];
#pragma unroll
    for (int ch = 0; ch < NCH; ++ch) {
        int c0 = ch * 4 + kadd;
        a2[ch].x = xr[Mr * 12 + c0];
        a2[ch].y = xr[Mr * 12 + c0 + 1];
    }
    int rbase = (lane < 16) ? 0 : 8;
    float sqr_r[8];
#pragma unroll
    for (int v = 0; v < 8; ++v) sqr_r[v] = sqr[v + rbase];   // hoisted: loop-invariant

    for (int ct = wv; ct < NP / 16; ct += 8) {
        int col = ct * 16 + Mr;
        const float* xc = x + (size_t)(b * NP + col) * 12;
        v8f c8 = {};
#pragma unroll
        for (int ch = 0; ch < NCH; ++ch) {
            int c0 = ch * 4 + kadd;
            v2f b2 = { xc[c0], xc[c0 + 1] };
            c8 = wmma_f32x4(a2[ch], b2, c8);
        }
        float sc = sq[col];
#pragma unroll
        for (int v = 0; v < 8; ++v) {
            float d2 = sqr_r[v] + sc - 2.0f * c8[v];
            dist[(v + rbase) * NPD + col] = (d2 > 0.0f) ? d2 : 0.0f;
        }
    }
    __syncthreads();

    // ---- Phase 2: top-(K+1) largest, drop first. Rows are wave-private
    // (wave w owns rows 2w, 2w+1), so only wave-level ordering is needed. ----
    int row = tid >> 4;      // 0..15
    int l16 = tid & 15;
    const float4* drow4 = (const float4*)(dist + row * NPD);
    for (int it = 0; it <= KNNK; ++it) {
        float best = -1.0f; int bidx = NP;
        for (int i = 0; i < NP / 64; ++i) {           // 64 x ds_load_b128
            float4 v4 = drow4[i * 16 + l16];
            int mb = i * 64 + l16 * 4;
            if (v4.x > best) { best = v4.x; bidx = mb; }
            if (v4.y > best) { best = v4.y; bidx = mb + 1; }
            if (v4.z > best) { best = v4.z; bidx = mb + 2; }
            if (v4.w > best) { best = v4.w; bidx = mb + 3; }
        }
        // argmax reduce over the 16-lane group (tie -> lowest index)
#pragma unroll
        for (int off = 8; off > 0; off >>= 1) {
            float ov = __shfl_down(best, off, 16);
            int   oi = __shfl_down(bidx, off, 16);
            if (ov > best || (ov == best && oi < bidx)) { best = ov; bidx = oi; }
        }
        if (l16 == 0) {
            dist[row * NPD + bidx] = -1.0f;
            if (it > 0) idx[(size_t)(b * NP + r0 + row) * KNNK + (it - 1)] = bidx;
        }
        wave_lds_fence();
    }
}

// ---------------------------------------------------------------------------
// Kernel 2: fused edge conv, templated on (D, EIN, CX, MODE).
// Block = 128 (4 waves), wave = 1 point (16 edges). All per-point LDS state is
// wave-private -> single block barrier (weight staging), wave fences elsewhere.
// ---------------------------------------------------------------------------
template <int D, int EIN, int CX, int MODE>
__global__ void egnn_edge_kernel(const float* __restrict__ x,
                                 const float* __restrict__ f,
                                 const int* __restrict__ idx,
                                 const float* __restrict__ We,
                                 const float* __restrict__ be,
                                 const float* __restrict__ Wx,
                                 const float* __restrict__ bx,
                                 float* __restrict__ xout,
                                 float* __restrict__ msum,
                                 float* __restrict__ outfinal) {
    constexpr int KCH = (EIN + 31) / 32;               // 1 or 5
    __shared__ __attribute__((aligned(64))) _Float16 WeP[KCH * 4 * 32 * 16];
    __shared__ __attribute__((aligned(64))) _Float16 featL[4][16 * 32];
    __shared__ float beL[64], WxL[CX * 64], bxL[CX];
    __shared__ float mL[4][16 * 64];
    __shared__ float xdL[4][16 * 12];
    __shared__ float dsqL[4][16];
    __shared__ float xcL[4][12];
    __shared__ float fcL[4][64];
    __shared__ float phiL[4][16 * 4];
    __shared__ int   nbL[4][16];

    int tid = threadIdx.x;

    // Pre-swizzled B: WeP[((ch*4+nt)*32+lane)*16 + t] = We[o=nt*16+(lane&15)]
    //                 [k = ch*32 + (lane<16?0:16) + t]  (zero-padded past EIN)
    for (int d = tid; d < KCH * 2048; d += 128) {
        int tq    = d & 15;
        int lane2 = (d >> 4) & 31;
        int nt    = (d >> 9) & 3;
        int ch    = d >> 11;
        int o     = nt * 16 + (lane2 & 15);
        int srck  = ch * 32 + ((lane2 < 16) ? 0 : 16) + tq;
        WeP[d] = (srck < EIN) ? (_Float16)We[o * EIN + srck] : (_Float16)0.0f;
    }
    if (tid < 64) beL[tid] = be[tid];
    for (int t = tid; t < CX * 64; t += 128) WxL[t] = Wx[t];
    if (tid < CX) bxL[tid] = bx[tid];

    int wv = tid >> 5, lane = tid & 31;
    int p = blockIdx.x * 4 + wv;
    int b = p / NP, n = p % NP;

    if (lane < KNNK) nbL[wv][lane] = idx[(size_t)(b * NP + n) * KNNK + lane];
    if (lane < 12)   xcL[wv][lane] = x[(size_t)(b * NP + n) * 12 + lane];
    for (int j = lane; j < 64; j += 32) fcL[wv][j] = f[(size_t)(b * NP + n) * 64 + j];
    __syncthreads();   // weight staging visible to all waves; per-wave state too

    if (lane < KNNK) {
        int nb = nbL[wv][lane];
        const float* xn = x + (size_t)(b * NP + nb) * 12;
        float s = 0.0f;
#pragma unroll
        for (int c = 0; c < 12; ++c) {
            float d = xn[c] - xcL[wv][c];      // zero-padded beyond Cc
            xdL[wv][lane * 12 + c] = d;
            s += d * d;
        }
        dsqL[wv][lane] = s;
    }
    wave_lds_fence();

    // ---- edge-MLP GEMM ----
    int Mr = lane & 15;
    int kb8 = (lane < 16) ? 0 : 1;   // A: half-waves hold K {0..15}/{16..31}
    v8f acc0 = {}, acc1 = {}, acc2 = {}, acc3 = {};
#pragma unroll
    for (int ch = 0; ch < KCH; ++ch) {
        {   // lane fills column j = lane of the 16x32 feat chunk
            int j  = lane;
            int jj = ch * 32 + j;
            for (int k = 0; k < KNNK; ++k) {
                float v;
                if (jj < D) {
                    int nb = nbL[wv][k];
                    v = f[(size_t)(b * NP + nb) * 64 + jj] - fcL[wv][jj];
                } else if (jj < 2 * D) {
                    v = fcL[wv][jj - D];
                } else if (jj == 2 * D) {
                    v = dsqL[wv][k];
                } else {
                    v = 0.0f;
                }
                featL[wv][k * 32 + j] = (_Float16)v;
            }
        }
        wave_lds_fence();
        const v8h* arow = (const v8h*)&featL[wv][Mr * 32];
        v16h a = cat8(arow[2 * kb8], arow[2 * kb8 + 1]);
        const v16h* bp = (const v16h*)&WeP[ch * 2048];
#pragma unroll
        for (int nt = 0; nt < 4; ++nt) {
            v16h bb = bp[nt * 32 + lane];
            if      (nt == 0) acc0 = wmma_f16(a, bb, acc0);
            else if (nt == 1) acc1 = wmma_f16(a, bb, acc1);
            else if (nt == 2) acc2 = wmma_f16(a, bb, acc2);
            else              acc3 = wmma_f16(a, bb, acc3);
        }
        wave_lds_fence();
    }
    int rbase = (lane < 16) ? 0 : 8;
#pragma unroll
    for (int nt = 0; nt < 4; ++nt) {
        v8f a8 = (nt == 0) ? acc0 : (nt == 1) ? acc1 : (nt == 2) ? acc2 : acc3;
        int coln = nt * 16 + Mr;
#pragma unroll
        for (int v = 0; v < 8; ++v) {
            float mv = silu_f(a8[v] + beL[coln]);
            mL[wv][(v + rbase) * 64 + coln] = mv;
        }
    }
    wave_lds_fence();

    // ---- msum ----
#pragma unroll
    for (int j = lane; j < 64; j += 32) {
        float s = 0.0f;
#pragma unroll
        for (int k = 0; k < KNNK; ++k) s += mL[wv][k * 64 + j];
        msum[(size_t)(b * NP + n) * 64 + j] = s;
    }

    // ---- phi ----
#pragma unroll
    for (int t = lane; t < KNNK * CX; t += 32) {
        int k = t / CX, c = t % CX;
        float s = bxL[c];
#pragma unroll
        for (int j = 0; j < 64; ++j) s += WxL[c * 64 + j] * mL[wv][k * 64 + j];
        phiL[wv][k * 4 + c] = silu_f(s);
    }
    wave_lds_fence();

    // ---- coordinate update ----
    const float inv_k = 1.0f / (float)KNNK;
    if constexpr (MODE == 0) {            // layer 1: broadcast x(3) over C=4
        if (lane < 12) {
            int c = lane / 3, d = lane % 3;
            float s = 0.0f;
#pragma unroll
            for (int k = 0; k < KNNK; ++k)
                s += xdL[wv][k * 12 + d] * phiL[wv][k * 4 + c];
            xout[(size_t)(b * NP + n) * 12 + lane] = xcL[wv][d] + s * inv_k;
        }
    } else if constexpr (MODE == 1) {     // mid layers
        if (lane < 12) {
            int c = lane / 3;
            float s = 0.0f;
#pragma unroll
            for (int k = 0; k < KNNK; ++k)
                s += xdL[wv][k * 12 + lane] * phiL[wv][k * 4 + c];
            xout[(size_t)(b * NP + n) * 12 + lane] = xcL[wv][lane] + s * inv_k;
        }
    } else {                              // final: fold C=4 -> C_OUT=2, transpose out
        if (lane < 6) {
            int c2 = lane / 3, d = lane % 3;
            float xe = 0.5f * (xcL[wv][c2 * 6 + d] + xcL[wv][c2 * 6 + 3 + d]);
            float s = 0.0f;
#pragma unroll
            for (int k = 0; k < KNNK; ++k) {
                float xde = 0.5f * (xdL[wv][k * 12 + c2 * 6 + d] +
                                    xdL[wv][k * 12 + c2 * 6 + 3 + d]);
                s += xde * phiL[wv][k * 4 + c2];
            }
            outfinal[(size_t)(b * 6 + lane) * NP + n] = xe + s * inv_k;
        }
    }
}

// ---------------------------------------------------------------------------
// Kernel 3: fmlp, templated on (D, F_IN).
// f_out = silu([fin(D), msum(64)] @ Wf^T + bf), rows = B*N.
// Block = 128 (4 waves), wave = 16 rows; pre-swizzled B like the edge kernel.
// ---------------------------------------------------------------------------
template <int D, int FIN>
__global__ void egnn_fmlp_kernel(const float* __restrict__ fin,
                                 const float* __restrict__ msum,
                                 const float* __restrict__ Wf,
                                 const float* __restrict__ bf,
                                 float* __restrict__ fout) {
    constexpr int KCH = (FIN + 31) / 32;               // 3 or 4
    __shared__ __attribute__((aligned(64))) _Float16 WfP[KCH * 4 * 32 * 16];
    __shared__ __attribute__((aligned(64))) _Float16 AL[4][16 * 32];
    __shared__ float bfL[64];

    int tid = threadIdx.x;
    for (int d = tid; d < KCH * 2048; d += 128) {
        int tq    = d & 15;
        int lane2 = (d >> 4) & 31;
        int nt    = (d >> 9) & 3;
        int ch    = d >> 11;
        int o     = nt * 16 + (lane2 & 15);
        int srck  = ch * 32 + ((lane2 < 16) ? 0 : 16) + tq;
        WfP[d] = (srck < FIN) ? (_Float16)Wf[o * FIN + srck] : (_Float16)0.0f;
    }
    if (tid < 64) bfL[tid] = bf[tid];
    __syncthreads();

    int wv = tid >> 5, lane = tid & 31;
    int r0w = blockIdx.x * 64 + wv * 16;
    int Mr = lane & 15;
    int kb8 = (lane < 16) ? 0 : 1;
    v8f acc0 = {}, acc1 = {}, acc2 = {}, acc3 = {};

#pragma unroll
    for (int ch = 0; ch < KCH; ++ch) {
        {
            int j  = lane;
            int jj = ch * 32 + j;
            for (int k = 0; k < 16; ++k) {
                int r = r0w + k;
                float v;
                if (jj < D)             v = fin[(size_t)r * 64 + jj];
                else if (jj < D + 64)   v = msum[(size_t)r * 64 + (jj - D)];
                else                    v = 0.0f;
                AL[wv][k * 32 + j] = (_Float16)v;
            }
        }
        wave_lds_fence();
        const v8h* arow = (const v8h*)&AL[wv][Mr * 32];
        v16h a = cat8(arow[2 * kb8], arow[2 * kb8 + 1]);
        const v16h* bp = (const v16h*)&WfP[ch * 2048];
#pragma unroll
        for (int nt = 0; nt < 4; ++nt) {
            v16h bb = bp[nt * 32 + lane];
            if      (nt == 0) acc0 = wmma_f16(a, bb, acc0);
            else if (nt == 1) acc1 = wmma_f16(a, bb, acc1);
            else if (nt == 2) acc2 = wmma_f16(a, bb, acc2);
            else              acc3 = wmma_f16(a, bb, acc3);
        }
        wave_lds_fence();
    }
    int rbase = (lane < 16) ? 0 : 8;
#pragma unroll
    for (int nt = 0; nt < 4; ++nt) {
        v8f a8 = (nt == 0) ? acc0 : (nt == 1) ? acc1 : (nt == 2) ? acc2 : acc3;
        int coln = nt * 16 + Mr;
#pragma unroll
        for (int v = 0; v < 8; ++v) {
            int r = r0w + rbase + v;
            fout[(size_t)r * 64 + coln] = silu_f(a8[v] + bfL[coln]);
        }
    }
}

// ---------------------------------------------------------------------------
extern "C" void kernel_launch(void* const* d_in, const int* in_sizes, int n_in,
                              void* d_out, int out_size, void* d_ws, size_t ws_size,
                              hipStream_t stream) {
    (void)in_sizes; (void)n_in; (void)out_size; (void)ws_size;
    const float* pts = (const float*)d_in[0];
    // dict order per layer l: We, be, Wx, bx, Wf, bf at d_in[1 + 6*l + s]
    auto Wp = [&](int l, int s) { return (const float*)d_in[1 + l * 6 + s]; };

    float* ws   = (float*)d_ws;
    float* xA   = ws;                                   // [B,N,12]
    float* xB   = xA + (size_t)BQ * NP * 12;
    float* fA   = xB + (size_t)BQ * NP * 12;            // [B,N,64]
    float* fB   = fA + (size_t)BQ * NP * 64;
    float* msum = fB + (size_t)BQ * NP * 64;            // [B,N,64]
    int*   idxb = (int*)(msum + (size_t)BQ * NP * 64);  // [B,N,16]
    float* out  = (float*)d_out;

    size_t knn_lds = (size_t)(16 * NPD + NP + 16 * 12 + 16) * 4;

    egnn_init_kernel<<<(BQ * NP + 255) / 256, 256, 0, stream>>>(pts, xA, fA);

    // ---- layer 1 ----
    egnn_knn_kernel<3><<<BQ * NP / 16, 256, knn_lds, stream>>>(xA, idxb);
    egnn_edge_kernel<1, 3, 4, 0><<<BQ * NP / 4, 128, 0, stream>>>(
        xA, fA, idxb, Wp(0,0), Wp(0,1), Wp(0,2), Wp(0,3), xB, msum, out);
    egnn_fmlp_kernel<1, 65><<<BQ * NP / 64, 128, 0, stream>>>(
        fA, msum, Wp(0,4), Wp(0,5), fB);

    // ---- layer 2 ----
    egnn_knn_kernel<12><<<BQ * NP / 16, 256, knn_lds, stream>>>(xB, idxb);
    egnn_edge_kernel<64, 129, 4, 1><<<BQ * NP / 4, 128, 0, stream>>>(
        xB, fB, idxb, Wp(1,0), Wp(1,1), Wp(1,2), Wp(1,3), xA, msum, out);
    egnn_fmlp_kernel<64, 128><<<BQ * NP / 64, 128, 0, stream>>>(
        fB, msum, Wp(1,4), Wp(1,5), fA);

    // ---- layer 3 ----
    egnn_knn_kernel<12><<<BQ * NP / 16, 256, knn_lds, stream>>>(xA, idxb);
    egnn_edge_kernel<64, 129, 4, 1><<<BQ * NP / 4, 128, 0, stream>>>(
        xA, fA, idxb, Wp(2,0), Wp(2,1), Wp(2,2), Wp(2,3), xB, msum, out);
    egnn_fmlp_kernel<64, 128><<<BQ * NP / 64, 128, 0, stream>>>(
        fA, msum, Wp(2,4), Wp(2,5), fB);

    // ---- layer 4 (writes d_out directly, no fmlp needed) ----
    egnn_knn_kernel<12><<<BQ * NP / 16, 256, knn_lds, stream>>>(xB, idxb);
    egnn_edge_kernel<64, 129, 2, 2><<<BQ * NP / 4, 128, 0, stream>>>(
        xB, fB, idxb, Wp(3,0), Wp(3,1), Wp(3,2), Wp(3,3), xA, msum, out);
}
